// STAttnBlock_44547400794815
// MI455X (gfx1250) — matrix operand order repaired and checked by reading, not compiled, not run on hardware
//
#include <hip/hip_runtime.h>

// ---------------------------------------------------------------------------
// STAttnBlock for MI455X (gfx1250): GroupNorm -> QKV -> flash attention -> proj
// bf16 WMMA (v_wmma_f32_16x16x32_bf16) for all GEMMs, fp32 accumulate.
// K/V/activation tiles staged in LDS via global_load_async_to_lds_b128
// (ASYNCcnt path) instead of VGPR round-trips.
// ---------------------------------------------------------------------------

typedef __bf16 bf16_t;
typedef __attribute__((ext_vector_type(16))) __bf16 v16bf;
typedef __attribute__((ext_vector_type(8)))  __bf16 v8bf;
typedef __attribute__((ext_vector_type(8)))  float  v8f;

#define CC   256      // channels
#define NN   6912     // tokens = F*H*W
#define HWsz 2304     // H*W
#define FF   3
#define BB   2        // batch after chunk
#define GG   4        // groupnorm groups

// Async copy 16B global -> LDS, per-lane addresses. The INST_OFFSET immediate
// is added to BOTH the global and LDS addresses (ISA 08_async_tensor §4.4),
// so one base pair + offsets covers a contiguous chunk.
#define ASYNC_LD128(ldsp, gp, OFF)                                              \
  asm volatile("global_load_async_to_lds_b128 %0, %1, off offset:" #OFF         \
               :: "v"((unsigned)(uintptr_t)(ldsp)), "v"((const void*)(gp))      \
               : "memory")

#define WAIT_ASYNC0() asm volatile("s_wait_asynccnt 0x0" ::: "memory")

__device__ __forceinline__ v16bf cat8(v8bf lo, v8bf hi) {
  return __builtin_shufflevector(lo, hi, 0,1,2,3,4,5,6,7,8,9,10,11,12,13,14,15);
}

// A-fragment (16x32 bf16): lane half = lane>>4. Per ISA: lanes 0-15 hold K
// 0..7 & 16..23; lanes 16-31 hold K 8..15 & 24..31 of their M row.
__device__ __forceinline__ v16bf load_afrag_bf16(const bf16_t* rowbase, int half) {
  v8bf lo = *(const v8bf*)(rowbase + half * 8);
  v8bf hi = *(const v8bf*)(rowbase + 16 + half * 8);
  return cat8(lo, hi);
}

// B-fragment (32x16 bf16): lane l holds column N=l%16; lanes 0-15 hold K
// 0..7 & 16..23, lanes 16-31 hold K 8..15 & 24..31 (stride ld between K rows).
__device__ __forceinline__ v16bf load_bfrag(const bf16_t* p, int ld, int half) {
  v16bf r;
#pragma unroll
  for (int s = 0; s < 8; s++) {
    r[s] = p[(size_t)(half * 8 + s) * ld];
    r[8 + s] = p[(size_t)(16 + half * 8 + s) * ld];
  }
  return r;
}

// A-fragment from fp32 memory (weights), converted to bf16 on the fly.
__device__ __forceinline__ v16bf load_afrag_f32(const float* rowbase, int half) {
  float4 a0 = *(const float4*)(rowbase + half * 8);
  float4 a1 = *(const float4*)(rowbase + half * 8 + 4);
  float4 b0 = *(const float4*)(rowbase + 16 + half * 8);
  float4 b1 = *(const float4*)(rowbase + 16 + half * 8 + 4);
  v16bf r;
  r[0]=(bf16_t)a0.x; r[1]=(bf16_t)a0.y; r[2]=(bf16_t)a0.z; r[3]=(bf16_t)a0.w;
  r[4]=(bf16_t)a1.x; r[5]=(bf16_t)a1.y; r[6]=(bf16_t)a1.z; r[7]=(bf16_t)a1.w;
  r[8]=(bf16_t)b0.x; r[9]=(bf16_t)b0.y; r[10]=(bf16_t)b0.z; r[11]=(bf16_t)b0.w;
  r[12]=(bf16_t)b1.x; r[13]=(bf16_t)b1.y; r[14]=(bf16_t)b1.z; r[15]=(bf16_t)b1.w;
  return r;
}

__device__ __forceinline__ v8f wmma_bf16(v16bf a, v16bf b, v8f c) {
  return __builtin_amdgcn_wmma_f32_16x16x32_bf16(false, a, false, b, (short)0, c,
                                                 false, false);
}

__device__ __forceinline__ float rmax16(float v) {
  v = fmaxf(v, __shfl_xor(v, 1, 32));
  v = fmaxf(v, __shfl_xor(v, 2, 32));
  v = fmaxf(v, __shfl_xor(v, 4, 32));
  v = fmaxf(v, __shfl_xor(v, 8, 32));
  return v;
}
__device__ __forceinline__ float rsum16(float v) {
  v += __shfl_xor(v, 1, 32);
  v += __shfl_xor(v, 2, 32);
  v += __shfl_xor(v, 4, 32);
  v += __shfl_xor(v, 8, 32);
  return v;
}

// ---------------------------------------------------------------------------
// 1) GroupNorm stats: one block per (b, g); mean/rstd over 64ch x 3f x HW.
// ---------------------------------------------------------------------------
__global__ void k_gn_stats(const float* __restrict__ x, float* __restrict__ stats) {
  const int bg = blockIdx.x;
  const int bi = bg >> 2, g = bg & 3;
  float s = 0.f, s2 = 0.f;
  for (int i = threadIdx.x; i < 64 * NN; i += 256) {
    int cl = i / NN, pos = i - cl * NN;
    int c = g * 64 + cl;
    int f = pos / HWsz, hw = pos - f * HWsz;
    float v = x[((size_t)(f * BB + bi) * CC + c) * HWsz + hw];
    s += v; s2 += v * v;
  }
  __shared__ float sh[512];
  sh[threadIdx.x] = s; sh[256 + threadIdx.x] = s2;
  __syncthreads();
  for (int st = 128; st > 0; st >>= 1) {
    if (threadIdx.x < st) {
      sh[threadIdx.x] += sh[threadIdx.x + st];
      sh[256 + threadIdx.x] += sh[256 + threadIdx.x + st];
    }
    __syncthreads();
  }
  if (threadIdx.x == 0) {
    const float cnt = 64.f * (float)NN;
    float mu = sh[0] / cnt;
    float var = sh[256] / cnt - mu * mu;
    stats[bg * 2] = mu;
    stats[bg * 2 + 1] = rsqrtf(var + 1e-6f);
  }
}

// ---------------------------------------------------------------------------
// 2) Apply GN, write bf16 channel-major hn[bi][c][n].
// ---------------------------------------------------------------------------
__global__ void k_gn_apply(const float* __restrict__ x, const float* __restrict__ gw,
                           const float* __restrict__ gb, const float* __restrict__ stats,
                           bf16_t* __restrict__ hn) {
  size_t idx = (size_t)blockIdx.x * 256 + threadIdx.x;  // = bi*C*N + c*N + n
  int bi = (int)(idx / ((size_t)CC * NN));
  int rem = (int)(idx - (size_t)bi * CC * NN);
  int c = rem / NN, n = rem - c * NN;
  int g = c >> 6;
  int f = n / HWsz, hw = n - f * HWsz;
  float mu = stats[(bi * GG + g) * 2];
  float rstd = stats[(bi * GG + g) * 2 + 1];
  float sc = gw[c] * rstd;
  float sh = gb[c] - mu * sc;
  float v = x[((size_t)(f * BB + bi) * CC + c) * HWsz + hw] * sc + sh;
  hn[idx] = (bf16_t)v;
}

// ---------------------------------------------------------------------------
// 3) QKV GEMM: D[o,n] = W[o,c] * hn[c,n] + bias. hn tile staged in 64KB LDS
//    via async-to-LDS copies. q,v stored token-major [n][C]; k channel-major.
// ---------------------------------------------------------------------------
__global__ void k_qkv(const bf16_t* __restrict__ hn,
                      const float* __restrict__ wq, const float* __restrict__ bq,
                      const float* __restrict__ wk, const float* __restrict__ bk,
                      const float* __restrict__ wv, const float* __restrict__ bv,
                      bf16_t* __restrict__ qb, bf16_t* __restrict__ kb,
                      bf16_t* __restrict__ vb) {
  __shared__ bf16_t ldsB[CC * 128];  // [c][128 tokens], 64 KB
  const int nb = blockIdx.x;         // 54 token blocks of 128
  const int which = blockIdx.y;      // 0=q 1=k 2=v
  const int bi = blockIdx.z;

  // async stage: thread t copies channel row t (128 bf16 = 256 B = 16 x b128)
  {
    const bf16_t* g = hn + ((size_t)bi * CC + threadIdx.x) * NN + nb * 128;
    bf16_t* l = &ldsB[threadIdx.x * 128];
#pragma unroll
    for (int i = 0; i < 16; i++) ASYNC_LD128(l + i * 8, g + i * 8, 0);
  }
  WAIT_ASYNC0();
  __syncthreads();

  const float* W    = (which == 0) ? wq : (which == 1) ? wk : wv;
  const float* bias = (which == 0) ? bq : (which == 1) ? bk : bv;
  bf16_t* dst       = (which == 0) ? qb : (which == 1) ? kb : vb;

  const int wv_ = threadIdx.x >> 5, lane = threadIdx.x & 31;
  const int half = lane >> 4, l16 = lane & 15;

#pragma unroll
  for (int oi = 0; oi < 2; oi++) {
    const int ot = wv_ * 2 + oi;
    v8f acc[8];
#pragma unroll
    for (int nt = 0; nt < 8; nt++) acc[nt] = (v8f){};
#pragma unroll
    for (int c8 = 0; c8 < 8; c8++) {
      v16bf af = load_afrag_f32(W + (size_t)(ot * 16 + l16) * CC + c8 * 32, half);
#pragma unroll
      for (int nt = 0; nt < 8; nt++) {
        v16bf bf = load_bfrag(&ldsB[(c8 * 32) * 128 + nt * 16 + l16], 128, half);
        acc[nt] = wmma_bf16(af, bf, acc[nt]);
      }
    }
    float bvv[8];
#pragma unroll
    for (int r = 0; r < 8; r++) bvv[r] = bias[ot * 16 + r + 8 * half];
#pragma unroll
    for (int nt = 0; nt < 8; nt++) {
#pragma unroll
      for (int r = 0; r < 8; r++) {
        float val = acc[nt][r] + bvv[r];
        int o = ot * 16 + r + 8 * half;
        int n = nb * 128 + nt * 16 + l16;
        if (which == 1)
          dst[(size_t)bi * CC * NN + (size_t)o * NN + n] = (bf16_t)val;        // k: [C][n]
        else
          dst[(size_t)bi * NN * CC + (size_t)n * CC + o] = (bf16_t)val;        // q,v: [n][C]
      }
    }
  }
}

// ---------------------------------------------------------------------------
// 4) Flash attention: 8 waves x 16 queries per block; key blocks of 32.
//    K/V tiles staged via async-to-LDS; P transposed through wave-private LDS.
// ---------------------------------------------------------------------------
__global__ void k_attn(const bf16_t* __restrict__ qbuf, const bf16_t* __restrict__ kbuf,
                       const bf16_t* __restrict__ vbuf, bf16_t* __restrict__ obuf) {
  __shared__ bf16_t ldsK[CC * 32];      // [c][32 keys], 16 KB
  __shared__ bf16_t ldsV[32 * CC];      // [key][c],     16 KB
  __shared__ bf16_t ldsP[8 * 16 * 32];  // per-wave P staging, 8 KB

  const int bi = blockIdx.y;
  const int wv_ = threadIdx.x >> 5, lane = threadIdx.x & 31;
  const int half = lane >> 4, l16 = lane & 15;
  const int qb = blockIdx.x * 128 + wv_ * 16;

  // Q fragments (A-layout), held for the whole kernel.
  v16bf qf[8];
#pragma unroll
  for (int c8 = 0; c8 < 8; c8++)
    qf[c8] = load_afrag_bf16(qbuf + (size_t)bi * NN * CC + (size_t)(qb + l16) * CC + c8 * 32, half);

  v8f oacc[16];
#pragma unroll
  for (int t = 0; t < 16; t++) oacc[t] = (v8f){};
  float mrun[8], lrun[8];
#pragma unroll
  for (int r = 0; r < 8; r++) { mrun[r] = -3.0e38f; lrun[r] = 0.f; }

  const float kscale2 = 0.0625f * 1.44269504088896340736f;  // C^-0.5 * log2(e)

  for (int kblk = 0; kblk < NN / 32; kblk++) {
    __syncthreads();
    // async stage K: thread t copies channel row t (32 keys = 64 B)
    {
      const bf16_t* g = kbuf + ((size_t)bi * CC + threadIdx.x) * NN + kblk * 32;
      bf16_t* l = &ldsK[threadIdx.x * 32];
      ASYNC_LD128(l, g, 0);
      ASYNC_LD128(l, g, 16);
      ASYNC_LD128(l, g, 32);
      ASYNC_LD128(l, g, 48);
    }
    // async stage V: thread t -> key row t/8, 64-byte chunk t%8
    {
      const int row = threadIdx.x >> 3, ch = threadIdx.x & 7;
      const bf16_t* g =
          vbuf + (size_t)bi * NN * CC + (size_t)(kblk * 32 + row) * CC + ch * 32;
      bf16_t* l = &ldsV[row * CC + ch * 32];
      ASYNC_LD128(l, g, 0);
      ASYNC_LD128(l, g, 16);
      ASYNC_LD128(l, g, 32);
      ASYNC_LD128(l, g, 48);
    }
    if (kblk + 1 < NN / 32) {  // prefetch next K tile -> global_prefetch
      __builtin_prefetch(kbuf + ((size_t)bi * CC + threadIdx.x) * NN + (kblk + 1) * 32, 0, 1);
    }
    WAIT_ASYNC0();
    __syncthreads();

    // scores: S[16q x 32k] as two 16x16 tiles, K-loop over channels.
    v8f s0 = (v8f){}, s1 = (v8f){};
#pragma unroll
    for (int c8 = 0; c8 < 8; c8++) {
      v16bf k0 = load_bfrag(&ldsK[(c8 * 32) * 32 + 0 + l16], 32, half);
      v16bf k1 = load_bfrag(&ldsK[(c8 * 32) * 32 + 16 + l16], 32, half);
      s0 = wmma_bf16(qf[c8], k0, s0);
      s1 = wmma_bf16(qf[c8], k1, s1);
    }

    // online softmax (raw-score domain, scale folded into exp2)
    float alpha[8];
    v8f p0, p1;
#pragma unroll
    for (int r = 0; r < 8; r++) {
      float raw = rmax16(fmaxf(s0[r], s1[r]));
      float mnew = fmaxf(mrun[r], raw);
      alpha[r] = __builtin_exp2f((mrun[r] - mnew) * kscale2);
      mrun[r] = mnew;
      p0[r] = __builtin_exp2f((s0[r] - mnew) * kscale2);
      p1[r] = __builtin_exp2f((s1[r] - mnew) * kscale2);
      float rs = rsum16(p0[r] + p1[r]);
      lrun[r] = lrun[r] * alpha[r] + rs;
    }
#pragma unroll
    for (int t = 0; t < 16; t++) {
#pragma unroll
      for (int r = 0; r < 8; r++) oacc[t][r] *= alpha[r];
    }

    // transpose P (C-layout -> A-layout) via wave-private LDS staging
    bf16_t* pst = &ldsP[wv_ * 512];
#pragma unroll
    for (int r = 0; r < 8; r++) {
      const int row = r + 8 * half;
      pst[row * 32 + l16]      = (bf16_t)p0[r];
      pst[row * 32 + 16 + l16] = (bf16_t)p1[r];
    }
    v16bf pf = load_afrag_bf16(&pst[l16 * 32], half);

    // O += P * V
#pragma unroll
    for (int t = 0; t < 16; t++) {
      v16bf vf = load_bfrag(&ldsV[t * 16 + l16], CC, half);
      oacc[t] = wmma_bf16(pf, vf, oacc[t]);
    }
  }

  // normalize and store token-major bf16 o[bi][n][c]
  float inv[8];
#pragma unroll
  for (int r = 0; r < 8; r++) inv[r] = 1.0f / lrun[r];
#pragma unroll
  for (int t = 0; t < 16; t++) {
#pragma unroll
    for (int r = 0; r < 8; r++) {
      int n = qb + r + 8 * half;
      int c = t * 16 + l16;
      obuf[(size_t)bi * NN * CC + (size_t)n * CC + c] = (bf16_t)(oacc[t][r] * inv[r]);
    }
  }
}

// ---------------------------------------------------------------------------
// 5) Transpose wp (fp32 [o][c]) -> bf16 wpT [c][o]
// ---------------------------------------------------------------------------
__global__ void k_wpt(const float* __restrict__ wp, bf16_t* __restrict__ wpT) {
  int o = blockIdx.x, c = threadIdx.x;
  wpT[c * CC + o] = (bf16_t)wp[o * CC + c];
}

// ---------------------------------------------------------------------------
// 6) Projection + bias + residual, de-interleave frames into output layout.
//    outT[n][o] = sum_c o[n][c] * wpT[c][o]
// ---------------------------------------------------------------------------
__global__ void k_proj(const bf16_t* __restrict__ obuf, const bf16_t* __restrict__ wpT,
                       const float* __restrict__ bp, const float* __restrict__ x,
                       float* __restrict__ out) {
  const int bi = blockIdx.y;
  const int wv_ = threadIdx.x >> 5, lane = threadIdx.x & 31;
  const int half = lane >> 4, l16 = lane & 15;
  const int tb = blockIdx.x * 128 + wv_ * 16;

  v8f acc[16];
#pragma unroll
  for (int t = 0; t < 16; t++) acc[t] = (v8f){};

#pragma unroll
  for (int c8 = 0; c8 < 8; c8++) {
    v16bf af =
        load_afrag_bf16(obuf + (size_t)bi * NN * CC + (size_t)(tb + l16) * CC + c8 * 32, half);
#pragma unroll
    for (int ot = 0; ot < 16; ot++) {
      v16bf bf = load_bfrag(wpT + (size_t)(c8 * 32) * CC + ot * 16 + l16, CC, half);
      acc[ot] = wmma_bf16(af, bf, acc[ot]);
    }
  }

#pragma unroll
  for (int r = 0; r < 8; r++) {
    const int token = tb + r + 8 * half;
    const int f = token / HWsz, hw = token - f * HWsz;
#pragma unroll
    for (int ot = 0; ot < 16; ot++) {
      const int o = ot * 16 + l16;
      const size_t oidx = ((size_t)(bi * FF + f) * CC + o) * HWsz + hw;
      const size_t xidx = ((size_t)(f * BB + bi) * CC + o) * HWsz + hw;
      out[oidx] = x[xidx] + acc[ot][r] + bp[o];
    }
  }
}

// ---------------------------------------------------------------------------
extern "C" void kernel_launch(void* const* d_in, const int* in_sizes, int n_in,
                              void* d_out, int out_size, void* d_ws, size_t ws_size,
                              hipStream_t stream) {
  const float* x   = (const float*)d_in[0];
  const float* gnw = (const float*)d_in[1];
  const float* gnb = (const float*)d_in[2];
  const float* wq  = (const float*)d_in[3];
  const float* bq  = (const float*)d_in[4];
  const float* wk  = (const float*)d_in[5];
  const float* bk  = (const float*)d_in[6];
  const float* wv  = (const float*)d_in[7];
  const float* bv  = (const float*)d_in[8];
  const float* wp  = (const float*)d_in[9];
  const float* bp  = (const float*)d_in[10];
  float* out = (float*)d_out;

  char* ws = (char*)d_ws;
  const size_t SZ = (size_t)BB * CC * NN;  // bf16 elements per activation buffer
  float*  stats = (float*)ws;              // 16 floats, pad to 256 B
  bf16_t* hn  = (bf16_t*)(ws + 256);
  bf16_t* qb  = hn + SZ;
  bf16_t* kb  = qb + SZ;
  bf16_t* vb  = kb + SZ;
  bf16_t* ob  = vb + SZ;
  bf16_t* wpT = ob + SZ;                   // 256*256 bf16

  k_wpt<<<dim3(CC), dim3(CC), 0, stream>>>(wp, wpT);
  k_gn_stats<<<dim3(BB * GG), dim3(256), 0, stream>>>(x, stats);
  k_gn_apply<<<dim3((unsigned)(SZ / 256)), dim3(256), 0, stream>>>(x, gnw, gnb, stats, hn);
  k_qkv<<<dim3(NN / 128, 3, BB), dim3(256), 0, stream>>>(hn, wq, bq, wk, bk, wv, bv,
                                                         qb, kb, vb);
  k_attn<<<dim3(NN / 128, BB), dim3(256), 0, stream>>>(qb, kb, vb, ob);
  k_proj<<<dim3(NN / 128, BB), dim3(256), 0, stream>>>(ob, wpT, bp, x, out);
}
